// UserEncoder_16527034155209
// MI455X (gfx1250) — matrix operand order repaired and checked by reading, not compiled, not run on hardware
//
#include <hip/hip_runtime.h>
#include <hip/hip_bf16.h>

typedef __bf16 v16bf __attribute__((ext_vector_type(16)));
typedef float  v8f   __attribute__((ext_vector_type(8)));

static __device__ __forceinline__ __bf16 f2bf(float f){
  unsigned u = __builtin_bit_cast(unsigned, f);
  u = (u + 0x7FFFu + ((u >> 16) & 1u)) >> 16;           // round-to-nearest-even
  unsigned short s = (unsigned short)u;
  return __builtin_bit_cast(__bf16, s);
}
static __device__ __forceinline__ float bf2f_bits(unsigned short s){
  unsigned u = ((unsigned)s) << 16;
  return __builtin_bit_cast(float, u);
}
static __device__ __forceinline__ float sigmoidf_(float x){ return 1.0f/(1.0f+__expf(-x)); }

// A-matrix (16x32 bf16) per-lane K index for element e (matches B packing):
// lanes 0-15: K = {0..15}; lanes 16-31: K = {16..31}
static __device__ __forceinline__ int a_kidx(int lane, int e){
  int base = (lane < 16) ? 0 : 16;
  return base + e;
}

// ---------------- weight packing into WMMA B-tile order ----------------
// tile(kb,nt): lane L holds 16 contiguous bf16; n = nt*16 + (L%16),
// k = kb*32 + (L/16)*16 + i  (B 32x16: lanes 0-15 -> K 0..15, 16-31 -> K 16..31)
__global__ void pack_tiles_bf16(const float* __restrict__ src, __bf16* __restrict__ dst,
                                int Krows, int Ncols, int NT, int total,
                                int transpose, int src_ld){
  int o = blockIdx.x*256 + threadIdx.x;
  if (o >= total) return;
  int tile = o >> 9;
  int r    = o & 511;
  int lane = r >> 4;
  int i    = r & 15;
  int kb = tile / NT, nt = tile - kb*NT;
  int k = kb*32 + ((lane >> 4) << 4) + i;
  int n = nt*16 + (lane & 15);
  float v = 0.f;
  if (k < Krows && n < Ncols)
    v = transpose ? src[(size_t)n*src_ld + k] : src[(size_t)k*src_ld + n];
  dst[o] = f2bf(v);
}

// --------- fused embed-gather + conv1d(K=3)+ReLU + attention proj + scores ---------
// one block per (b,n) title; 8 waves. Each job = one N-tile, BOTH M-tiles (B reuse x2).
__global__ __launch_bounds__(256) void news_conv_attn_kernel(
    const int* __restrict__ title, const float* __restrict__ word_emb,
    const __bf16* __restrict__ convP, const float* __restrict__ conv_b,
    const __bf16* __restrict__ vP, const float* __restrict__ vb,
    const float* __restrict__ q,
    __bf16* __restrict__ Cout, float* __restrict__ scores)
{
  __shared__ __bf16 Xs[36][320];   // row0 = left pad, rows1..30 = words, rows31..35 pad
  __shared__ __bf16 Cs[32][416];   // conv output (bf16), K-padded to 416
  __shared__ float  Ts[32][208];   // tanh proj
  __shared__ int    tw[30];
  const int bn   = blockIdx.x;
  const int tid  = threadIdx.x;
  const int wave = tid >> 5, lane = tid & 31;
  const int ml   = lane & 15;

  if (tid < 30) tw[tid] = title[bn*30 + tid];
  __syncthreads();
  for (int i = tid; i < 36*320; i += 256){
    int row = i / 320, col = i - row*320;
    float v = 0.f;
    if (row >= 1 && row <= 30 && col < 300) v = word_emb[(size_t)tw[row-1]*300 + col];
    Xs[row][col] = f2bf(v);
  }
  for (int i = tid; i < 32*16; i += 256) Cs[i>>4][400 + (i&15)] = f2bf(0.f);
  __syncthreads();

  // conv as 3 shifted GEMMs: C[t,f] += X[t+tap-1,:] @ W[tap];  M=32, K=320(pad), N=400
  for (int nt = wave; nt < 25; nt += 8){
    v8f acc0 = {}, acc1 = {};
    for (int tap = 0; tap < 3; ++tap){
      for (int kb = 0; kb < 10; ++kb){
        v16bf a0, a1, b;
        #pragma unroll
        for (int e = 0; e < 16; ++e){
          int kk = kb*32 + a_kidx(lane, e);
          a0[e] = Xs[ml + tap][kk];
          a1[e] = Xs[16 + ml + tap][kk];
        }
        b = *(const v16bf*)(convP + (size_t)(((tap*10 + kb)*25 + nt) << 9) + lane*16);
        acc0 = __builtin_amdgcn_wmma_f32_16x16x32_bf16(false, a0, false, b, (short)0, acc0, false, false);
        acc1 = __builtin_amdgcn_wmma_f32_16x16x32_bf16(false, a1, false, b, (short)0, acc1, false, false);
      }
    }
    int col = nt*16 + ml;
    float bias = conv_b[col];
    #pragma unroll
    for (int r = 0; r < 8; ++r){
      int mrow = r + ((lane < 16) ? 0 : 8);
      float v0 = fmaxf(acc0[r] + bias, 0.f);
      float v1 = fmaxf(acc1[r] + bias, 0.f);
      __bf16 h0 = f2bf(v0), h1 = f2bf(v1);
      Cs[mrow][col] = h0;
      Cs[16 + mrow][col] = h1;
      Cout[((size_t)bn*30 + mrow)*400 + col] = h0;
      if (16 + mrow < 30) Cout[((size_t)bn*30 + 16 + mrow)*400 + col] = h1;
    }
  }
  __syncthreads();

  // tanh(C @ v + vb):  M=32, K=416(pad), N=208(pad)
  for (int nt = wave; nt < 13; nt += 8){
    v8f acc0 = {}, acc1 = {};
    for (int kb = 0; kb < 13; ++kb){
      v16bf a0, a1, b;
      #pragma unroll
      for (int e = 0; e < 16; ++e){
        int kk = kb*32 + a_kidx(lane, e);
        a0[e] = Cs[ml][kk];
        a1[e] = Cs[16 + ml][kk];
      }
      b = *(const v16bf*)(vP + (size_t)((kb*13 + nt) << 9) + lane*16);
      acc0 = __builtin_amdgcn_wmma_f32_16x16x32_bf16(false, a0, false, b, (short)0, acc0, false, false);
      acc1 = __builtin_amdgcn_wmma_f32_16x16x32_bf16(false, a1, false, b, (short)0, acc1, false, false);
    }
    int col = nt*16 + ml;
    float bias = (col < 200) ? vb[col] : 0.f;
    #pragma unroll
    for (int r = 0; r < 8; ++r){
      int mrow = r + ((lane < 16) ? 0 : 8);
      Ts[mrow][col] = tanhf(acc0[r] + bias);
      Ts[16 + mrow][col] = tanhf(acc1[r] + bias);
    }
  }
  __syncthreads();

  if (tid < 30){
    float s = 0.f;
    for (int a2 = 0; a2 < 200; ++a2) s += Ts[tid][a2] * q[a2];
    scores[bn*30 + tid] = s;
  }
}

// softmax over the N(titles) axis for each (b,t)
__global__ void softmax_titles_kernel(const float* __restrict__ scores, float* __restrict__ alpha){
  int idx = blockIdx.x*256 + threadIdx.x;
  if (idx >= 64*30) return;
  int b = idx / 30, t = idx - b*30;
  const float* s = scores + b*1500 + t;
  float mx = -1e30f;
  for (int n = 0; n < 50; ++n) mx = fmaxf(mx, s[n*30]);
  float sum = 0.f;
  for (int n = 0; n < 50; ++n) sum += __expf(s[n*30] - mx);
  float inv = 1.f / sum;
  float* a = alpha + b*1500 + t;
  for (int n = 0; n < 50; ++n) a[n*30] = __expf(s[n*30] - mx) * inv;
}

// e[bn,f] = sum_t alpha[bn,t] * C[bn,t,f] ; each thread handles a pair of f (u32 bf16x2 loads)
__global__ void weighted_sum_kernel(const float* __restrict__ alpha,
                                    const unsigned* __restrict__ C2,   // C as bf16x2
                                    float* __restrict__ ne){
  int idx = blockIdx.x*256 + threadIdx.x;        // over 3200*200 pairs
  if (idx >= 3200*200) return;
  int bn = idx / 200, fp = idx - bn*200;
  float s0 = 0.f, s1 = 0.f;
  for (int t = 0; t < 30; ++t){
    float av = alpha[bn*30 + t];
    unsigned u = C2[((size_t)bn*30 + t)*200 + fp];
    s0 += av * bf2f_bits((unsigned short)(u & 0xFFFFu));
    s1 += av * bf2f_bits((unsigned short)(u >> 16));
  }
  ne[bn*400 + fp*2]     = s0;
  ne[bn*400 + fp*2 + 1] = s1;
}

// GX = news_embed @ W_ih^T + b_ih : M=3200, K=400, N=1200
// one block per M-tile; A staged in LDS once, each wave loops over N-tiles
__global__ __launch_bounds__(256) void gx_gemm_kernel(const float* __restrict__ ne,
    const __bf16* __restrict__ WihP, const float* __restrict__ b_ih, float* __restrict__ GX){
  __shared__ __bf16 As[16][416];
  const int tid = threadIdx.x;
  const int wave = tid >> 5, lane = tid & 31, ml = lane & 15;
  const int m0 = blockIdx.x * 16;
  for (int i = tid; i < 16*416; i += 256){
    int row = i / 416, col = i - row*416;
    float v = (col < 400) ? ne[(size_t)(m0 + row)*400 + col] : 0.f;
    As[row][col] = f2bf(v);
  }
  __syncthreads();
  for (int nt = wave; nt < 75; nt += 8){
    v8f acc = {};
    for (int kb = 0; kb < 13; ++kb){
      v16bf a, b;
      #pragma unroll
      for (int e = 0; e < 16; ++e)
        a[e] = As[ml][kb*32 + a_kidx(lane, e)];
      b = *(const v16bf*)(WihP + (size_t)((kb*75 + nt) << 9) + lane*16);
      acc = __builtin_amdgcn_wmma_f32_16x16x32_bf16(false, a, false, b, (short)0, acc, false, false);
    }
    int n = nt*16 + ml;
    float bias = b_ih[n];
    #pragma unroll
    for (int r = 0; r < 8; ++r){
      int mrow = m0 + r + ((lane < 16) ? 0 : 8);
      GX[(size_t)mrow*1200 + n] = acc[r] + bias;
    }
  }
}

__global__ void h0_init_kernel(const int* __restrict__ user_id, const float* __restrict__ user_emb,
                               float* __restrict__ h){
  int idx = blockIdx.x*256 + threadIdx.x;
  if (idx >= 64*400) return;
  int b = idx / 400, g = idx - b*400;
  h[idx] = user_emb[(size_t)user_id[b]*400 + g];
}

// one GRU step: gh = h @ W_hh^T (3 gates via WMMA) + fused pointwise update
// grid = 25 (hidden 16-col tiles), block = 128 (4 waves = 4 row tiles of 16 -> 64 rows)
// full h staged in LDS once per block, shared by all waves
__global__ __launch_bounds__(128) void gru_step_kernel(const float* __restrict__ h_in,
    float* __restrict__ h_out, const float* __restrict__ GX,
    const __bf16* __restrict__ WhhP, const float* __restrict__ b_hh,
    const int* __restrict__ lengths, int step)
{
  __shared__ __bf16 Hs[64][416];
  const int tid = threadIdx.x;
  const int wave = tid >> 5, lane = tid & 31, ml = lane & 15;
  const int jb = blockIdx.x;
  const int m0 = wave*16;
  for (int i = tid; i < 64*416; i += 128){
    int row = i / 416, col = i - row*416;
    float v = (col < 400) ? h_in[row*400 + col] : 0.f;
    Hs[row][col] = f2bf(v);
  }
  __syncthreads();
  v8f ar = {}, az = {}, an = {};
  for (int kb = 0; kb < 13; ++kb){
    v16bf a;
    #pragma unroll
    for (int e = 0; e < 16; ++e)
      a[e] = Hs[m0 + ml][kb*32 + a_kidx(lane, e)];
    v16bf br  = *(const v16bf*)(WhhP + (size_t)((kb*75 +      jb) << 9) + lane*16);
    v16bf bz  = *(const v16bf*)(WhhP + (size_t)((kb*75 + 25 + jb) << 9) + lane*16);
    v16bf bn2 = *(const v16bf*)(WhhP + (size_t)((kb*75 + 50 + jb) << 9) + lane*16);
    ar = __builtin_amdgcn_wmma_f32_16x16x32_bf16(false, a, false, br,  (short)0, ar, false, false);
    az = __builtin_amdgcn_wmma_f32_16x16x32_bf16(false, a, false, bz,  (short)0, az, false, false);
    an = __builtin_amdgcn_wmma_f32_16x16x32_bf16(false, a, false, bn2, (short)0, an, false, false);
  }
  int j = jb*16 + ml;
  float bhr = b_hh[j], bhz = b_hh[400 + j], bhn = b_hh[800 + j];
  #pragma unroll
  for (int r = 0; r < 8; ++r){
    int m = m0 + r + ((lane < 16) ? 0 : 8);
    const float* gx = GX + ((size_t)m*50 + step)*1200 + j;
    float rr = sigmoidf_(gx[0]   + ar[r] + bhr);
    float zz = sigmoidf_(gx[400] + az[r] + bhz);
    float nn = tanhf(gx[800] + rr*(an[r] + bhn));
    float hp = h_in[m*400 + j];
    float hn = (1.f - zz)*nn + zz*hp;
    h_out[m*400 + j] = (step < lengths[m]) ? hn : hp;
  }
}

extern "C" void kernel_launch(void* const* d_in, const int* in_sizes, int n_in,
                              void* d_out, int out_size, void* d_ws, size_t ws_size,
                              hipStream_t stream){
  (void)in_sizes; (void)n_in; (void)out_size; (void)ws_size;
  const int*   user_id  = (const int*)d_in[0];
  const int*   title    = (const int*)d_in[1];
  const int*   lengths  = (const int*)d_in[2];
  const float* word_emb = (const float*)d_in[3];
  const float* conv_w   = (const float*)d_in[4];
  const float* conv_b   = (const float*)d_in[5];
  const float* v        = (const float*)d_in[6];
  const float* vb       = (const float*)d_in[7];
  const float* q        = (const float*)d_in[8];
  const float* user_emb = (const float*)d_in[9];
  const float* W_ih     = (const float*)d_in[10];
  const float* W_hh     = (const float*)d_in[11];
  const float* b_ih     = (const float*)d_in[12];
  const float* b_hh     = (const float*)d_in[13];
  float* out = (float*)d_out;

  char* w = (char*)d_ws;
  size_t off = 0;
  auto alloc = [&](size_t bytes){ void* p = w + off; off = (off + bytes + 255) & ~(size_t)255; return p; };
  __bf16* convP = (__bf16*)alloc(3u*10*25*512*2);   // 768 KB
  __bf16* vP    = (__bf16*)alloc(13u*13*512*2);
  __bf16* WihP  = (__bf16*)alloc(13u*75*512*2);
  __bf16* WhhP  = (__bf16*)alloc(13u*75*512*2);
  __bf16* Cbuf  = (__bf16*)alloc((size_t)3200*30*400*2);  // 76.8 MB
  float*  scores= (float*)alloc(96000u*4);
  float*  alpha = (float*)alloc(96000u*4);
  float*  ne    = (float*)alloc((size_t)3200*400*4);
  float*  GX    = (float*)alloc((size_t)3200*1200*4);
  float*  hA    = (float*)alloc(64u*400*4);
  float*  hB    = (float*)alloc(64u*400*4);

  for (int tap = 0; tap < 3; ++tap){
    int total = 10*25*512;
    pack_tiles_bf16<<<(total+255)/256, 256, 0, stream>>>(conv_w + (size_t)tap*300*400,
        convP + (size_t)tap*10*25*512, 300, 400, 25, total, 0, 400);
  }
  { int total = 13*13*512; pack_tiles_bf16<<<(total+255)/256,256,0,stream>>>(v,    vP,   400,  200, 13, total, 0, 200); }
  { int total = 13*75*512; pack_tiles_bf16<<<(total+255)/256,256,0,stream>>>(W_ih, WihP, 400, 1200, 75, total, 1, 400); }
  { int total = 13*75*512; pack_tiles_bf16<<<(total+255)/256,256,0,stream>>>(W_hh, WhhP, 400, 1200, 75, total, 1, 400); }

  news_conv_attn_kernel<<<3200, 256, 0, stream>>>(title, word_emb, convP, conv_b, vP, vb, q, Cbuf, scores);
  softmax_titles_kernel<<<8, 256, 0, stream>>>(scores, alpha);
  weighted_sum_kernel<<<2500, 256, 0, stream>>>(alpha, (const unsigned*)Cbuf, ne);
  gx_gemm_kernel<<<200, 256, 0, stream>>>(ne, WihP, b_ih, GX);
  h0_init_kernel<<<100, 256, 0, stream>>>(user_id, user_emb, hA);

  float* cur = hA; float* oth = hB;
  for (int s = 0; s < 50; ++s){
    float* dst = (s == 49) ? out : oth;
    gru_step_kernel<<<25, 128, 0, stream>>>(cur, dst, GX, WhhP, b_hh, lengths, s);
    float* t = cur; cur = oth; oth = t;
  }
}
